// HierarchicalFusionExpert_19043884990741
// MI455X (gfx1250) — matrix operand order repaired and checked by reading, not compiled, not run on hardware
//
#include <hip/hip_runtime.h>

// ---------------------------------------------------------------------------
// MI455X (gfx1250): all GEMMs via v_wmma_f32_16x16x32_bf16, f32 accumulate.
// B matrices staged TRANSPOSED (N x K) so every WMMA fragment is loaded from
// LDS with contiguous ds_load_b128 (2 per A frag, 2 per B frag) -- no u16
// gathers. Compute-bound workload (~240 GFLOP, AI >> 10 FLOP/byte).
// ---------------------------------------------------------------------------

typedef __bf16 bf16_t;
typedef bf16_t v16bf __attribute__((ext_vector_type(16)));
typedef float  v8f   __attribute__((ext_vector_type(8)));

union FragB16 { v16bf v; unsigned short u[16]; uint4 q[2]; };
union FragF32 { v8f   v; float          f[8];  };

__device__ __forceinline__ unsigned short f2bf(float f) {
    unsigned int u = __float_as_uint(f);
    unsigned int r = u + 0x7FFFu + ((u >> 16) & 1u);   // round-to-nearest-even
    return (unsigned short)(r >> 16);
}

// ---------------------------------------------------------------------------
// Batched bf16 GEMM: C[M,N] = A[M,K](bf16) @ B[K,N] + bias, with B given
// TRANSPOSED as Bt[N,K](bf16). Tile 128x128x32, 256 threads = 8 waves, each
// wave owns 32x64 (2x4 WMMA tiles). Optional f32 and/or bf16 outputs.
// Requires M%128==0, N%128==0, K%32==0 (true for every GEMM here).
// ---------------------------------------------------------------------------
#define TM 128
#define TN 128
#define TK 32
#define LDT 48   // LDS row stride in shorts (96B: 16B-aligned, bank-spread)

__global__ __launch_bounds__(256) void k_gemm_bf16(
    const unsigned short* __restrict__ A,
    const unsigned short* __restrict__ Bt,
    float* __restrict__ Cf,
    unsigned short* __restrict__ Cbf,
    const float* __restrict__ bias,
    int M, int N, int K,
    long long sA, long long sB, long long sC)
{
    __shared__ unsigned short As[TM * LDT];   // 12 KB  (A  tile: [m][k])
    __shared__ unsigned short Bs[TN * LDT];   // 12 KB  (Bt tile: [n][k])

    const int tid  = threadIdx.x;
    const int bz   = blockIdx.z;
    const unsigned short* Ab = A  + (long long)bz * sA;
    const unsigned short* Bb = Bt + (long long)bz * sB;
    const int m0 = blockIdx.y * TM;
    const int n0 = blockIdx.x * TN;

    const int wave = tid >> 5;
    const int lane = tid & 31;
    const int half = lane >> 4;
    const int l16  = lane & 15;
    const int wm   = (wave & 3) * 32;   // wave row offset within tile
    const int wn   = (wave >> 2) * 64;  // wave col offset within tile

    // global->LDS addressing: 512 16B chunks per tile, 2 per thread
    const int rowA = tid >> 2;           // 0..63  (+64 for second chunk)
    const int cgA  = (tid & 3) << 3;     // k-group *8 shorts

    FragF32 acc[2][4];
#pragma unroll
    for (int i = 0; i < 2; ++i)
#pragma unroll
        for (int j = 0; j < 4; ++j)
#pragma unroll
            for (int r = 0; r < 8; ++r) acc[i][j].f[r] = 0.0f;

    for (int k0 = 0; k0 < K; k0 += TK) {
        if (k0 + TK < K) {   // emits global_prefetch_b8
            __builtin_prefetch((const void*)(Ab + (long long)(m0 + rowA) * K + k0 + TK), 0, 0);
            __builtin_prefetch((const void*)(Bb + (long long)(n0 + rowA) * K + k0 + TK), 0, 0);
        }
#pragma unroll
        for (int r = 0; r < 2; ++r) {
            int row = rowA + r * 64;
            *reinterpret_cast<uint4*>(&As[row * LDT + cgA]) =
                *reinterpret_cast<const uint4*>(Ab + (long long)(m0 + row) * K + k0 + cgA);
            *reinterpret_cast<uint4*>(&Bs[row * LDT + cgA]) =
                *reinterpret_cast<const uint4*>(Bb + (long long)(n0 + row) * K + k0 + cgA);
        }
        __syncthreads();

        // A frag (16x32 bf16, ISA 7.12.2): lane(h,m) = rows of two 16B chunks
        FragB16 af[2];
#pragma unroll
        for (int i = 0; i < 2; ++i) {
            const unsigned short* ap = &As[(wm + i * 16 + l16) * LDT];
            af[i].q[0] = *reinterpret_cast<const uint4*>(ap + half * 8);
            af[i].q[1] = *reinterpret_cast<const uint4*>(ap + 16 + half * 8);
        }
        // B frag (32x16): lane(h,n) holds K=8h..8h+7 and 16+8h..16+8h+7 of
        // column n -> two 16B chunks of Bt row n (same order as A)
        FragB16 bf[4];
#pragma unroll
        for (int j = 0; j < 4; ++j) {
            const unsigned short* bp = &Bs[(wn + j * 16 + l16) * LDT];
            bf[j].q[0] = *reinterpret_cast<const uint4*>(bp + half * 8);
            bf[j].q[1] = *reinterpret_cast<const uint4*>(bp + 16 + half * 8);
        }
#pragma unroll
        for (int i = 0; i < 2; ++i)
#pragma unroll
            for (int j = 0; j < 4; ++j)
                acc[i][j].v = __builtin_amdgcn_wmma_f32_16x16x32_bf16(
                    false, af[i].v, false, bf[j].v, (short)0, acc[i][j].v, false, false);
        __syncthreads();
    }

    // C/D f32 16x16 layout: VGPR r -> row = r + 8*half, col = lane%16
#pragma unroll
    for (int i = 0; i < 2; ++i)
#pragma unroll
        for (int j = 0; j < 4; ++j)
#pragma unroll
            for (int r = 0; r < 8; ++r) {
                int row = m0 + wm + i * 16 + half * 8 + r;
                int col = n0 + wn + j * 16 + l16;
                float v = acc[i][j].f[r];
                if (bias) v += bias[col];
                long long o = (long long)bz * sC + (long long)row * N + col;
                if (Cf)  Cf[o]  = v;
                if (Cbf) Cbf[o] = f2bf(v);
            }
}

// ---------------------------------------------------------------------------
// Tiled transpose + f32->bf16: dst[N,K] = bf16(src[K,N]^T). grid.z batches.
// ---------------------------------------------------------------------------
__global__ __launch_bounds__(256) void k_transpose_cvt(
    const float* __restrict__ src, unsigned short* __restrict__ dst, int K, int N)
{
    __shared__ float tile[32][33];
    const long long boff = (long long)blockIdx.z * K * N;
    const int k0 = blockIdx.y * 32, n0 = blockIdx.x * 32;
    const int tx = threadIdx.x & 31, ty = threadIdx.x >> 5;
    for (int r = ty; r < 32; r += 8)
        tile[r][tx] = src[boff + (long long)(k0 + r) * N + n0 + tx];
    __syncthreads();
    for (int r = ty; r < 32; r += 8)
        dst[boff + (long long)(n0 + r) * K + k0 + tx] = f2bf(tile[tx][r]);
}

// ---------------------------------------------------------------------------
// Fused LayerNorm + exact-erf GELU; one block per row. Optional bf16 output
// and/or f32 accumulation (qubit-sum path).
// ---------------------------------------------------------------------------
__global__ __launch_bounds__(256) void k_ln_gelu(
    const float* __restrict__ in, const float* __restrict__ g,
    const float* __restrict__ b, unsigned short* __restrict__ outb,
    float* __restrict__ accp, int firstAcc, int Dd)
{
    const int row = blockIdx.x;
    const int t   = threadIdx.x;
    const float* x = in + (long long)row * Dd;
    __shared__ float red[256];
    const float invD = 1.0f / (float)Dd;

    float s = 0.0f;
    for (int d = t; d < Dd; d += 256) s += x[d];
    red[t] = s; __syncthreads();
    for (int o = 128; o > 0; o >>= 1) { if (t < o) red[t] += red[t + o]; __syncthreads(); }
    const float mean = red[0] * invD;
    __syncthreads();

    float v = 0.0f;
    for (int d = t; d < Dd; d += 256) { float z = x[d] - mean; v += z * z; }
    red[t] = v; __syncthreads();
    for (int o = 128; o > 0; o >>= 1) { if (t < o) red[t] += red[t + o]; __syncthreads(); }
    const float rstd = rsqrtf(red[0] * invD + 1e-5f);
    __syncthreads();

    for (int d = t; d < Dd; d += 256) {
        float y = (x[d] - mean) * rstd * g[d] + b[d];
        y = 0.5f * y * (1.0f + erff(y * 0.70710678118654752f));   // exact GELU
        long long o = (long long)row * Dd + d;
        if (outb) outb[o] = f2bf(y);
        if (accp) accp[o] = firstAcc ? y : (accp[o] + y);
    }
}

// mixed = (sum_q qs) * mean(mixer,0) / Q  ->  bf16
__global__ __launch_bounds__(256) void k_scale_mixed(
    const float* __restrict__ accp, const float* __restrict__ mixer,
    unsigned short* __restrict__ outb, long long n, int Dd, float invQ3)
{
    for (long long i = blockIdx.x * (long long)blockDim.x + threadIdx.x; i < n;
         i += (long long)gridDim.x * blockDim.x) {
        int d = (int)(i % Dd);
        float mm = (mixer[d] + mixer[Dd + d] + mixer[2 * Dd + d]) * invQ3;
        outb[i] = f2bf(accp[i] * mm);
    }
}

__global__ __launch_bounds__(256) void k_cvt_bf16(
    const float* __restrict__ in, unsigned short* __restrict__ out, long long n)
{
    for (long long i = blockIdx.x * (long long)blockDim.x + threadIdx.x; i < n;
         i += (long long)gridDim.x * blockDim.x) out[i] = f2bf(in[i]);
}

// qkv [B,L,3,H,HD] f32 -> Q(bf16, pre-scaled, [BH,L,HD]),
//                         K(bf16, [BH,L,HD])  (== Bt for S = Q K^T),
//                         V^T(bf16, [BH,HD,L]) (== Bt for O = P V)
__global__ __launch_bounds__(256) void k_split_qkv(
    const float* __restrict__ qkv, unsigned short* __restrict__ Qb,
    unsigned short* __restrict__ Kn, unsigned short* __restrict__ Vt,
    int Bn, int Ln, int Hn, int HDn, float qscale)
{
    long long n = (long long)Bn * Ln * Hn * HDn;
    for (long long i = blockIdx.x * (long long)blockDim.x + threadIdx.x; i < n;
         i += (long long)gridDim.x * blockDim.x) {
        int hd = (int)(i % HDn); long long tl = i / HDn;
        int h = (int)(tl % Hn);  tl /= Hn;
        int l = (int)(tl % Ln);  int b = (int)(tl / Ln);
        long long base = (((long long)(b * Ln + l) * 3) * Hn + h) * HDn + hd;
        long long bh = (long long)b * Hn + h;
        Qb[(bh * Ln + l) * HDn + hd] = f2bf(qkv[base] * qscale);
        Kn[(bh * Ln + l) * HDn + hd] = f2bf(qkv[base + (long long)Hn * HDn]);
        Vt[(bh * HDn + hd) * Ln + l] = f2bf(qkv[base + 2LL * Hn * HDn]);
    }
}

// Causal eps-softmax over one score row; writes bf16 probabilities.
__global__ __launch_bounds__(256) void k_softmax_causal(
    const float* __restrict__ S, unsigned short* __restrict__ P, int Ln)
{
    const int q  = blockIdx.x;
    const int bh = blockIdx.y;
    const int t  = threadIdx.x;
    const float* s = S + ((long long)bh * Ln + q) * (long long)Ln;
    unsigned short* p = P + ((long long)bh * Ln + q) * (long long)Ln;
    __shared__ float red[256];

    float mx = -__builtin_inff();
    for (int j = t; j < Ln; j += 256) if (j <= q) mx = fmaxf(mx, s[j]);
    red[t] = mx; __syncthreads();
    for (int o = 128; o > 0; o >>= 1) { if (t < o) red[t] = fmaxf(red[t], red[t + o]); __syncthreads(); }
    mx = red[0]; __syncthreads();

    float sum = 0.0f;
    for (int j = t; j < Ln; j += 256) if (j <= q) sum += __expf(s[j] - mx);
    red[t] = sum; __syncthreads();
    for (int o = 128; o > 0; o >>= 1) { if (t < o) red[t] += red[t + o]; __syncthreads(); }
    const float inv = 1.0f / (red[0] + 1e-6f);

    for (int j = t; j < Ln; j += 256)
        p[j] = f2bf((j <= q) ? __expf(s[j] - mx) * inv : 0.0f);
}

// O [BH,L,HD] f32 -> [B,L,D] bf16
__global__ __launch_bounds__(256) void k_merge_heads(
    const float* __restrict__ O, unsigned short* __restrict__ out,
    int Bn, int Ln, int Hn, int HDn)
{
    long long n = (long long)Bn * Ln * Hn * HDn;
    for (long long i = blockIdx.x * (long long)blockDim.x + threadIdx.x; i < n;
         i += (long long)gridDim.x * blockDim.x) {
        int hd = (int)(i % HDn); long long tl = i / HDn;
        int h = (int)(tl % Hn);  tl /= Hn;
        int l = (int)(tl % Ln);  int b = (int)(tl / Ln);
        float v = O[(((long long)b * Hn + h) * Ln + l) * HDn + hd];
        out[((long long)(b * Ln + l)) * (Hn * HDn) + h * HDn + hd] = f2bf(v);
    }
}

// level_out = softmax(lm[level]) . (qo, ao); total (+)= level_out; cur = bf16
__global__ __launch_bounds__(256) void k_combine(
    const float* __restrict__ qo, const float* __restrict__ ao,
    const float* __restrict__ lm, int level, float* __restrict__ total,
    unsigned short* __restrict__ cur, long long n, int first)
{
    float a = lm[level * 2], b = lm[level * 2 + 1];
    float m = fmaxf(a, b);
    float ea = __expf(a - m), eb = __expf(b - m);
    float w0 = ea / (ea + eb), w1 = eb / (ea + eb);
    for (long long i = blockIdx.x * (long long)blockDim.x + threadIdx.x; i < n;
         i += (long long)gridDim.x * blockDim.x) {
        float lo = w0 * qo[i] + w1 * ao[i];
        total[i] = first ? lo : (total[i] + lo);
        cur[i] = f2bf(lo);
    }
}

// ---------------------------------------------------------------------------
// Host orchestration
// ---------------------------------------------------------------------------
extern "C" void kernel_launch(void* const* d_in, const int* in_sizes, int n_in,
                              void* d_out, int out_size, void* d_ws, size_t ws_size,
                              hipStream_t stream)
{
    const int Bn = 2, Ln = 1024, Dd = 1024, CP = 256, Hn = 8, HDn = 128, Qn = 8;
    const int Mr = Bn * Ln;          // 2048 GEMM rows
    const int BH = Bn * Hn;          // 16 attention batches
    const int LEVELS = 3;
    const int EW = 2048;             // elementwise grid

    // pytree (sorted dict key) flattening: level_mixing, then per level
    // [attn: dec1.W,dec1.b,dec_ln.b,dec_ln.g,enc1.W,enc1.b,enc2.W,enc2.b,
    //        enc_ln.b,enc_ln.g,out.W,out.b,proj.W,proj.b,qkv.W,qkv.b]
    // [quantum: mixer,post1.W,post1.b,post2.W,post2.b,post_ln.b,post_ln.g,
    //           pre.W,pre.b,pre_ln.b,pre_ln.g,qW,q_ln_b,q_ln_g,qb], then x.
    const float* lm = (const float*)d_in[0];
    const float* x  = (const float*)d_in[n_in - 1];
    float* totalOut = (float*)d_out;

    size_t off = 0;
    auto alloc = [&](size_t bytes) -> void* {
        void* p = (char*)d_ws + off;
        off += (bytes + 255) & ~(size_t)255;
        return p;
    };
    // bf16 TRANSPOSED weight staging (reused per level): W^T is [N][K]
    unsigned short* wPre   = (unsigned short*)alloc((size_t)Dd * Dd * 2);
    unsigned short* wQ     = (unsigned short*)alloc((size_t)Qn * Dd * Dd * 2);
    unsigned short* wPost1 = (unsigned short*)alloc((size_t)Dd * Dd * 2);
    unsigned short* wPost2 = (unsigned short*)alloc((size_t)Dd * Dd * 2);
    unsigned short* wEnc1  = (unsigned short*)alloc((size_t)Dd * Dd * 2);
    unsigned short* wEnc2  = (unsigned short*)alloc((size_t)Dd * CP * 2);
    unsigned short* wDec1  = (unsigned short*)alloc((size_t)CP * Dd * 2);
    unsigned short* wQkv   = (unsigned short*)alloc((size_t)Dd * 3 * Dd * 2);
    unsigned short* wProj  = (unsigned short*)alloc((size_t)Dd * Dd * 2);
    unsigned short* wOut   = (unsigned short*)alloc((size_t)Dd * Dd * 2);
    // activations
    unsigned short* curB  = (unsigned short*)alloc((size_t)Mr * Dd * 2);
    unsigned short* hB    = (unsigned short*)alloc((size_t)Mr * Dd * 2);
    unsigned short* mixB  = (unsigned short*)alloc((size_t)Mr * Dd * 2);
    unsigned short* cB    = (unsigned short*)alloc((size_t)Mr * Dd * 2);
    unsigned short* compB = (unsigned short*)alloc((size_t)Mr * CP * 2);
    float* tmpA = (float*)alloc((size_t)Mr * 3 * Dd * 4);
    float* accF = (float*)alloc((size_t)Mr * Dd * 4);
    float* qoF  = (float*)alloc((size_t)Mr * Dd * 4);
    float* aoF  = (float*)alloc((size_t)Mr * Dd * 4);
    unsigned short* Qb = (unsigned short*)alloc((size_t)BH * Ln * HDn * 2);
    unsigned short* Kn = (unsigned short*)alloc((size_t)BH * Ln * HDn * 2);
    unsigned short* Vt = (unsigned short*)alloc((size_t)BH * HDn * Ln * 2);
    float*          Sf = (float*)alloc((size_t)BH * Ln * Ln * 4);
    unsigned short* Pb = (unsigned short*)alloc((size_t)BH * Ln * Ln * 2);
    float*          Of = (float*)alloc((size_t)BH * Ln * HDn * 4);
    if (off > ws_size) return;   // workspace too small: bail deterministically

    auto cvtT = [&](const float* src, unsigned short* dst, int K, int N, int batch) {
        dim3 g(N / 32, K / 32, batch);
        k_transpose_cvt<<<g, 256, 0, stream>>>(src, dst, K, N);
    };
    auto gemm = [&](const unsigned short* A, const unsigned short* Bt, float* Cf,
                    unsigned short* Cbf, const float* bias, int M, int N, int K,
                    int batch, long long sA, long long sB, long long sC) {
        dim3 g(N / TN, M / TM, batch);
        k_gemm_bf16<<<g, 256, 0, stream>>>(A, Bt, Cf, Cbf, bias, M, N, K, sA, sB, sC);
    };
    auto lngelu = [&](const float* in, const float* g, const float* b,
                      unsigned short* outb, float* accp, int first) {
        k_ln_gelu<<<Mr, 256, 0, stream>>>(in, g, b, outb, accp, first, Dd);
    };

    k_cvt_bf16<<<EW, 256, 0, stream>>>(x, curB, (long long)Mr * Dd);  // current = x

    for (int lvl = 0; lvl < LEVELS; ++lvl) {
        const int LB = 1 + lvl * 31;
        auto f = [&](int k) { return (const float*)d_in[LB + k]; };
        // attn leaves
        const float *dec1W = f(0),  *dec1b = f(1),  *declnB = f(2), *declnG = f(3);
        const float *enc1W = f(4),  *enc1b = f(5),  *enc2W = f(6),  *enc2b = f(7);
        const float *enclnB = f(8), *enclnG = f(9);
        const float *outW = f(10),  *outb = f(11),  *projW = f(12), *projb = f(13);
        const float *qkvW = f(14),  *qkvb = f(15);
        // quantum leaves
        const float *mixer = f(16);
        const float *post1W = f(17), *post1b = f(18), *post2W = f(19), *post2b = f(20);
        const float *postlnB = f(21), *postlnG = f(22);
        const float *preW = f(23), *preb = f(24), *prelnB = f(25), *prelnG = f(26);
        const float *qW = f(27), *qlnb = f(28), *qlng = f(29), *qb = f(30);

        // stage this level's weights transposed as bf16
        cvtT(preW,   wPre,   Dd, Dd, 1);
        cvtT(qW,     wQ,     Dd, Dd, Qn);
        cvtT(post1W, wPost1, Dd, Dd, 1);
        cvtT(post2W, wPost2, Dd, Dd, 1);
        cvtT(enc1W,  wEnc1,  Dd, CP * 0 + Dd, 1);
        cvtT(enc2W,  wEnc2,  Dd, CP, 1);
        cvtT(dec1W,  wDec1,  CP, Dd, 1);
        cvtT(qkvW,   wQkv,   Dd, 3 * Dd, 1);
        cvtT(projW,  wProj,  Dd, Dd, 1);
        cvtT(outW,   wOut,   Dd, Dd, 1);

        // ---- quantum expert ----
        gemm(curB, wPre, tmpA, nullptr, preb, Mr, Dd, Dd, 1, 0, 0, 0);
        lngelu(tmpA, prelnG, prelnB, hB, nullptr, 0);
        for (int q = 0; q < Qn; ++q) {
            gemm(hB, wQ + (long long)q * Dd * Dd, tmpA, nullptr, qb + q * Dd,
                 Mr, Dd, Dd, 1, 0, 0, 0);
            lngelu(tmpA, qlng + q * Dd, qlnb + q * Dd, nullptr, accF, q == 0);
        }
        k_scale_mixed<<<EW, 256, 0, stream>>>(accF, mixer, mixB,
                                              (long long)Mr * Dd, Dd, 1.0f / (3.0f * Qn));
        gemm(mixB, wPost1, tmpA, nullptr, post1b, Mr, Dd, Dd, 1, 0, 0, 0);
        lngelu(tmpA, postlnG, postlnB, hB, nullptr, 0);
        gemm(hB, wPost2, qoF, nullptr, post2b, Mr, Dd, Dd, 1, 0, 0, 0);

        // ---- attention expert ----
        gemm(curB, wEnc1, tmpA, nullptr, enc1b, Mr, Dd, Dd, 1, 0, 0, 0);
        lngelu(tmpA, enclnG, enclnB, cB, nullptr, 0);
        gemm(cB, wEnc2, nullptr, compB, enc2b, Mr, CP, Dd, 1, 0, 0, 0);  // fused bf16 out
        gemm(compB, wDec1, tmpA, nullptr, dec1b, Mr, Dd, CP, 1, 0, 0, 0);
        lngelu(tmpA, declnG, declnB, cB, nullptr, 0);
        gemm(cB, wQkv, tmpA, nullptr, qkvb, Mr, 3 * Dd, Dd, 1, 0, 0, 0);
        k_split_qkv<<<EW, 256, 0, stream>>>(tmpA, Qb, Kn, Vt, Bn, Ln, Hn, HDn,
                                            rsqrtf((float)HDn));
        // S = (Q*scale) @ K^T : Bt == K in natural [L][HD] layout
        gemm(Qb, Kn, Sf, nullptr, nullptr, Ln, Ln, HDn, BH,
             (long long)Ln * HDn, (long long)Ln * HDn, (long long)Ln * Ln);
        k_softmax_causal<<<dim3(Ln, BH), 256, 0, stream>>>(Sf, Pb, Ln);
        // O = P @ V : Bt == V^T [HD][L]
        gemm(Pb, Vt, Of, nullptr, nullptr, Ln, HDn, Ln, BH,
             (long long)Ln * Ln, (long long)HDn * Ln, (long long)Ln * HDn);
        k_merge_heads<<<EW, 256, 0, stream>>>(Of, cB, Bn, Ln, Hn, HDn);
        gemm(cB, wProj, nullptr, hB, projb, Mr, Dd, Dd, 1, 0, 0, 0);     // fused bf16 out
        gemm(hB, wOut, aoF, nullptr, outb, Mr, Dd, Dd, 1, 0, 0, 0);

        // ---- level mixing ----
        k_combine<<<EW, 256, 0, stream>>>(qoF, aoF, lm, lvl, totalOut, curB,
                                          (long long)Mr * Dd, lvl == 0);
    }
    (void)in_sizes; (void)out_size;
}
